// MultiheadSelfAttention_63239098466756
// MI455X (gfx1250) — matrix operand order repaired and checked by reading, not compiled, not run on hardware
//
#include <hip/hip_runtime.h>

// ---------------------------------------------------------------------------
// Types
// ---------------------------------------------------------------------------
typedef __attribute__((ext_vector_type(16))) __bf16          bf16x16;
typedef __attribute__((ext_vector_type(8)))  float           f32x8;
typedef __attribute__((ext_vector_type(4)))  float           f32x4;
typedef __attribute__((ext_vector_type(16))) unsigned short  u16x16;
typedef __attribute__((ext_vector_type(8)))  unsigned short  u16x8;

#define DEV static __device__ __forceinline__

// Problem constants
constexpr int Bc = 4, Sc = 2048, Dc = 1024, Hc = 16, HDc = 64;
constexpr int Mrows = Bc * Sc;          // 8192
constexpr int Nqkv  = 3 * Dc;           // 3072

// ---------------------------------------------------------------------------
// bf16 helpers (manual RNE so we control the hi/lo split exactly)
// ---------------------------------------------------------------------------
DEV unsigned short bf16_rne(float x) {
    unsigned u = __builtin_bit_cast(unsigned, x);
    u += 0x7FFFu + ((u >> 16) & 1u);
    return (unsigned short)(u >> 16);
}
DEV float bf16_up(unsigned short h) {
    unsigned u = ((unsigned)h) << 16;
    return __builtin_bit_cast(float, u);
}
DEV void bf16_split(float x, unsigned short &hi, unsigned short &lo) {
    hi = bf16_rne(x);
    lo = bf16_rne(x - bf16_up(hi));
}

DEV u16x8 ld8(const unsigned short *p, unsigned off) {
    return *reinterpret_cast<const u16x8 *>(p + off);
}
DEV bf16x16 mkfrag(u16x8 a, u16x8 b) {
    u16x16 t = __builtin_shufflevector(a, b, 0,1,2,3,4,5,6,7,8,9,10,11,12,13,14,15);
    return __builtin_bit_cast(bf16x16, t);
}
DEV f32x8 wmma_bf16(bf16x16 a, bf16x16 b, f32x8 c) {
    // D = A*B + C, 16x16x32 bf16 -> f32 accum
    return __builtin_amdgcn_wmma_f32_16x16x32_bf16(false, a, false, b, (short)0, c, false, false);
}

// ---------------------------------------------------------------------------
// Kernel 0a: elementwise split f32 -> bf16 hi/lo planes (same layout)
// ---------------------------------------------------------------------------
__global__ __launch_bounds__(256) void k_split(const float *__restrict__ src,
                                               unsigned short *__restrict__ hi,
                                               unsigned short *__restrict__ lo,
                                               int n) {
    unsigned i = blockIdx.x * 256 + threadIdx.x;
    if (i >= (unsigned)n) return;
    unsigned short h, l;
    bf16_split(src[i], h, l);
    hi[i] = h;
    lo[i] = l;
}

// ---------------------------------------------------------------------------
// Kernel 0b: transpose + split: W[K x N] row-major -> Wt_hi/lo[N x K]
// ---------------------------------------------------------------------------
__global__ __launch_bounds__(256) void k_transpose_split(const float *__restrict__ W,
                                                         unsigned short *__restrict__ hiT,
                                                         unsigned short *__restrict__ loT,
                                                         int K, int N) {
    unsigned i = blockIdx.x * 256 + threadIdx.x;
    if (i >= (unsigned)(K * N)) return;
    unsigned k = i / (unsigned)N, n = i - k * (unsigned)N;
    unsigned short h, l;
    bf16_split(W[i], h, l);
    unsigned o = n * (unsigned)K + k;
    hiT[o] = h;
    loT[o] = l;
}

// ---------------------------------------------------------------------------
// Shared GEMM main loop: C(128x128 per block) = A(M x 1024) * B^T(N x 1024)
// A, B stored as bf16 hi/lo planes, K = lda = ldb = 1024.
// Block: 256 threads = 8 waves in 4x2; each wave: 2 mtiles x 4 ntiles of 16x16.
// 3-term split product: hi*hi + hi*lo + lo*hi.
// All offsets kept in 32-bit so the backend can use SADDR+voffset addressing.
// ---------------------------------------------------------------------------
DEV void gemm_tiles(const unsigned short *__restrict__ Ahi,
                    const unsigned short *__restrict__ Alo,
                    const unsigned short *__restrict__ Bhi,
                    const unsigned short *__restrict__ Blo,
                    int m0, int n0, int ln, int grp, f32x8 acc[2][4]) {
    const unsigned aRow0 = (unsigned)(m0 + ln) * 1024u + (unsigned)(grp * 8);
    const unsigned aRow1 = aRow0 + 16u * 1024u;
    const unsigned bCol  = (unsigned)(n0 + ln) * 1024u + (unsigned)(grp * 8);
    for (unsigned k0 = 0; k0 < 1024; k0 += 32) {
        bf16x16 ah[2], al[2];
        {
            unsigned b0 = aRow0 + k0, b1 = aRow1 + k0;
            ah[0] = mkfrag(ld8(Ahi, b0), ld8(Ahi, b0 + 16));
            al[0] = mkfrag(ld8(Alo, b0), ld8(Alo, b0 + 16));
            ah[1] = mkfrag(ld8(Ahi, b1), ld8(Ahi, b1 + 16));
            al[1] = mkfrag(ld8(Alo, b1), ld8(Alo, b1 + 16));
        }
#pragma unroll
        for (int nt = 0; nt < 4; ++nt) {
            unsigned base = bCol + (unsigned)(nt * 16) * 1024u + k0;
            bf16x16 bh = mkfrag(ld8(Bhi, base), ld8(Bhi, base + 16));
            bf16x16 bl = mkfrag(ld8(Blo, base), ld8(Blo, base + 16));
#pragma unroll
            for (int mt = 0; mt < 2; ++mt) {
                acc[mt][nt] = wmma_bf16(ah[mt], bh, acc[mt][nt]);
                acc[mt][nt] = wmma_bf16(ah[mt], bl, acc[mt][nt]);
                acc[mt][nt] = wmma_bf16(al[mt], bh, acc[mt][nt]);
            }
        }
    }
}

// ---------------------------------------------------------------------------
// Kernel 1: QKV projection.  Out: Q,K head-major [B,H,S,HD] bf16 hi/lo
// (Q pre-scaled by 1/8), V transposed [B,H,HD,S] bf16 hi/lo.
// ---------------------------------------------------------------------------
__global__ __launch_bounds__(256) void k_qkv_gemm(
    const unsigned short *__restrict__ Xhi, const unsigned short *__restrict__ Xlo,
    const unsigned short *__restrict__ Wth, const unsigned short *__restrict__ Wtl,
    const float *__restrict__ bqkv,
    unsigned short *__restrict__ Qhi, unsigned short *__restrict__ Qlo,
    unsigned short *__restrict__ Khi, unsigned short *__restrict__ Klo,
    unsigned short *__restrict__ Vthi, unsigned short *__restrict__ Vtlo) {
    const int lane = threadIdx.x & 31, w = threadIdx.x >> 5;
    const int ln = lane & 15, grp = lane >> 4;
    const int m0 = blockIdx.y * 128 + (w >> 1) * 32;
    const int n0 = blockIdx.x * 128 + (w & 1) * 64;

    f32x8 acc[2][4] = {};
    gemm_tiles(Xhi, Xlo, Wth, Wtl, m0, n0, ln, grp, acc);

#pragma unroll
    for (int nt = 0; nt < 4; ++nt) {
        int gcol = n0 + nt * 16 + ln;
        float bias = bqkv[gcol];
        int h = gcol / 192;
        int sub = gcol - h * 192; // 0..191; q:0-63 k:64-127 v:128-191
#pragma unroll
        for (int mt = 0; mt < 2; ++mt) {
#pragma unroll
            for (int v = 0; v < 8; ++v) {
                int row = m0 + mt * 16 + grp * 8 + v;   // C frag: VGPR v, half grp
                int b = row >> 11, s = row & 2047;
                float val = acc[mt][nt][v] + bias;
                unsigned short hh, ll;
                if (sub < 64) {
                    val *= 0.125f; // 1/sqrt(64) folded into Q
                    bf16_split(val, hh, ll);
                    unsigned o = ((unsigned)(b * Hc + h) * Sc + s) * HDc + sub;
                    Qhi[o] = hh; Qlo[o] = ll;
                } else if (sub < 128) {
                    bf16_split(val, hh, ll);
                    unsigned o = ((unsigned)(b * Hc + h) * Sc + s) * HDc + (sub - 64);
                    Khi[o] = hh; Klo[o] = ll;
                } else {
                    bf16_split(val, hh, ll);
                    unsigned o = ((unsigned)(b * Hc + h) * HDc + (sub - 128)) * Sc + s;
                    Vthi[o] = hh; Vtlo[o] = ll;
                }
            }
        }
    }
}

// ---------------------------------------------------------------------------
// Kernel 2: flash attention forward.
// Block = 128 threads (4 waves); each wave owns 16 query rows, streams 64-key
// tiles up to val_len. Writes O as bf16 hi/lo planes, layout [B,S, H*HD].
// ---------------------------------------------------------------------------
__global__ __launch_bounds__(128) void k_attn(
    const unsigned short *__restrict__ Qhi, const unsigned short *__restrict__ Qlo,
    const unsigned short *__restrict__ Khi, const unsigned short *__restrict__ Klo,
    const unsigned short *__restrict__ Vthi, const unsigned short *__restrict__ Vtlo,
    const int *__restrict__ val_lens,
    unsigned short *__restrict__ Ohi, unsigned short *__restrict__ Olo) {
    constexpr int PST = 68; // padded LDS row stride (floats) -> conflict-free b128
    __shared__ float Pbuf[4][16 * PST];

    const int lane = threadIdx.x & 31, w = threadIdx.x >> 5;
    const int ln = lane & 15, grp = lane >> 4;
    const int b = blockIdx.z, h = blockIdx.y;
    const int q0 = blockIdx.x * 64 + w * 16;
    const int vl = val_lens[b];
    const int nkv = (vl + 63) >> 6;

    const unsigned qkBase = (unsigned)(b * Hc + h) * Sc * HDc;  // [S][HD]
    const unsigned vtBase = (unsigned)(b * Hc + h) * HDc * Sc;  // [HD][S]
    float *P = &Pbuf[w][0];

    // Q fragments (A-layout), reused across all kv tiles
    bf16x16 qh[2], ql[2];
#pragma unroll
    for (int kd = 0; kd < 2; ++kd) {
        unsigned base = qkBase + (unsigned)(q0 + ln) * HDc + kd * 32 + grp * 8;
        qh[kd] = mkfrag(ld8(Qhi, base), ld8(Qhi, base + 16));
        ql[kd] = mkfrag(ld8(Qlo, base), ld8(Qlo, base + 16));
    }

    f32x8 o[4] = {};
    float mrow[8], lrow[8];
#pragma unroll
    for (int v = 0; v < 8; ++v) { mrow[v] = -3.0e38f; lrow[v] = 0.0f; }

    for (int t = 0; t < nkv; ++t) {
        const int kb = t * 64;
        // ----- scores: S = Q * K^T  (pre-scaled Q) -----
        f32x8 sacc[4] = {};
#pragma unroll
        for (int nt = 0; nt < 4; ++nt) {
            unsigned kybase = qkBase + (unsigned)(kb + nt * 16 + ln) * HDc;
#pragma unroll
            for (int kd = 0; kd < 2; ++kd) {
                unsigned off = kybase + kd * 32 + grp * 8;
                bf16x16 bh = mkfrag(ld8(Khi, off), ld8(Khi, off + 16));
                bf16x16 bl = mkfrag(ld8(Klo, off), ld8(Klo, off + 16));
                sacc[nt] = wmma_bf16(qh[kd], bh, sacc[nt]);
                sacc[nt] = wmma_bf16(qh[kd], bl, sacc[nt]);
                sacc[nt] = wmma_bf16(ql[kd], bh, sacc[nt]);
            }
        }
        // ----- key-length mask (columns = lanes) -----
        if (kb + 64 > vl) {
#pragma unroll
            for (int nt = 0; nt < 4; ++nt) {
                bool dead = (kb + nt * 16 + ln) >= vl;
#pragma unroll
                for (int v = 0; v < 8; ++v)
                    sacc[nt][v] = dead ? -1000000.0f : sacc[nt][v];
            }
        }
        // ----- online softmax update -----
        float rm[8];
#pragma unroll
        for (int v = 0; v < 8; ++v) {
            float a = fmaxf(sacc[0][v], sacc[1][v]);
            float c = fmaxf(sacc[2][v], sacc[3][v]);
            rm[v] = fmaxf(a, c);
#pragma unroll
            for (int off = 1; off <= 8; off <<= 1)
                rm[v] = fmaxf(rm[v], __shfl_xor(rm[v], off));
        }
        float rs[8];
#pragma unroll
        for (int v = 0; v < 8; ++v) {
            float nm = fmaxf(mrow[v], rm[v]);
            float sc = __expf(mrow[v] - nm);
            mrow[v] = nm;
            lrow[v] *= sc;
            rs[v] = 0.0f;
#pragma unroll
            for (int nt = 0; nt < 4; ++nt) o[nt][v] *= sc;
        }
#pragma unroll
        for (int nt = 0; nt < 4; ++nt) {
#pragma unroll
            for (int v = 0; v < 8; ++v) {
                float p = __expf(sacc[nt][v] - mrow[v]);
                rs[v] += p;
                P[(grp * 8 + v) * PST + nt * 16 + ln] = p;
            }
        }
#pragma unroll
        for (int v = 0; v < 8; ++v) {
#pragma unroll
            for (int off = 1; off <= 8; off <<= 1)
                rs[v] += __shfl_xor(rs[v], off);
            lrow[v] += rs[v];
        }
        asm volatile("s_wait_dscnt 0" ::: "memory"); // wave-local LDS RAW
        // ----- O += P * V -----
#pragma unroll
        for (int kd = 0; kd < 2; ++kd) {
            // P A-fragment from LDS (f32 -> bf16 hi/lo split here)
            const float *pr = &P[ln * PST + kd * 32 + grp * 8];
            f32x4 a0 = *(const f32x4 *)(pr);
            f32x4 a1 = *(const f32x4 *)(pr + 4);
            f32x4 a2 = *(const f32x4 *)(pr + 16);
            f32x4 a3 = *(const f32x4 *)(pr + 20);
            u16x16 th, tl;
#pragma unroll
            for (int i = 0; i < 4; ++i) {
                unsigned short hh, ll;
                bf16_split(a0[i], hh, ll); th[i]      = hh; tl[i]      = ll;
                bf16_split(a1[i], hh, ll); th[4 + i]  = hh; tl[4 + i]  = ll;
                bf16_split(a2[i], hh, ll); th[8 + i]  = hh; tl[8 + i]  = ll;
                bf16_split(a3[i], hh, ll); th[12 + i] = hh; tl[12 + i] = ll;
            }
            bf16x16 ph = __builtin_bit_cast(bf16x16, th);
            bf16x16 pl = __builtin_bit_cast(bf16x16, tl);
#pragma unroll
            for (int nt = 0; nt < 4; ++nt) {
                unsigned voff = vtBase + (unsigned)(nt * 16 + ln) * Sc + kb + kd * 32 + grp * 8;
                bf16x16 vh = mkfrag(ld8(Vthi, voff), ld8(Vthi, voff + 16));
                bf16x16 vo = mkfrag(ld8(Vtlo, voff), ld8(Vtlo, voff + 16));
                o[nt] = wmma_bf16(ph, vh, o[nt]);
                o[nt] = wmma_bf16(ph, vo, o[nt]);
                o[nt] = wmma_bf16(pl, vh, o[nt]);
            }
        }
    }

    // ----- finalize: O /= l, write [B,S,H*HD] bf16 hi/lo -----
#pragma unroll
    for (int nt = 0; nt < 4; ++nt) {
#pragma unroll
        for (int v = 0; v < 8; ++v) {
            int row = q0 + grp * 8 + v;
            float val = o[nt][v] / lrow[v];
            unsigned short hh, ll;
            bf16_split(val, hh, ll);
            unsigned oo = ((unsigned)(b * Sc + row)) * Dc + h * HDc + nt * 16 + ln;
            Ohi[oo] = hh;
            Olo[oo] = ll;
        }
    }
}

// ---------------------------------------------------------------------------
// Kernel 3: output projection, f32 result + bias
// ---------------------------------------------------------------------------
__global__ __launch_bounds__(256) void k_out_gemm(
    const unsigned short *__restrict__ Ohi, const unsigned short *__restrict__ Olo,
    const unsigned short *__restrict__ Wth, const unsigned short *__restrict__ Wtl,
    const float *__restrict__ bout, float *__restrict__ out) {
    const int lane = threadIdx.x & 31, w = threadIdx.x >> 5;
    const int ln = lane & 15, grp = lane >> 4;
    const int m0 = blockIdx.y * 128 + (w >> 1) * 32;
    const int n0 = blockIdx.x * 128 + (w & 1) * 64;

    f32x8 acc[2][4] = {};
    gemm_tiles(Ohi, Olo, Wth, Wtl, m0, n0, ln, grp, acc);

#pragma unroll
    for (int nt = 0; nt < 4; ++nt) {
        int gcol = n0 + nt * 16 + ln;
        float bias = bout[gcol];
#pragma unroll
        for (int mt = 0; mt < 2; ++mt) {
#pragma unroll
            for (int v = 0; v < 8; ++v) {
                int row = m0 + mt * 16 + grp * 8 + v;
                out[(unsigned)row * Dc + gcol] = acc[mt][nt][v] + bias;
            }
        }
    }
}

// ---------------------------------------------------------------------------
// Host launcher
// ---------------------------------------------------------------------------
extern "C" void kernel_launch(void *const *d_in, const int *in_sizes, int n_in,
                              void *d_out, int out_size, void *d_ws, size_t ws_size,
                              hipStream_t stream) {
    const float *x    = (const float *)d_in[0];
    const float *Wqkv = (const float *)d_in[1];
    const float *bqkv = (const float *)d_in[2];
    const float *Wout = (const float *)d_in[3];
    const float *bout = (const float *)d_in[4];
    const int   *vlen = (const int *)d_in[5];
    float *out = (float *)d_out;

    // Workspace layout (bytes)
    unsigned char *p = (unsigned char *)d_ws;
    constexpr size_t XPLANE = (size_t)Mrows * Dc * 2;        // 16 MiB
    constexpr size_t WQT    = (size_t)Nqkv * Dc * 2;         // 6 MiB
    constexpr size_t WOT    = (size_t)Dc * Dc * 2;           // 2 MiB
    constexpr size_t HPLANE = (size_t)Bc * Hc * Sc * HDc * 2;// 16 MiB
    size_t off = 0;
    unsigned short *Xhi  = (unsigned short *)(p + off); off += XPLANE;
    unsigned short *Xlo  = (unsigned short *)(p + off); off += XPLANE;
    unsigned short *Wqth = (unsigned short *)(p + off); off += WQT;
    unsigned short *Wqtl = (unsigned short *)(p + off); off += WQT;
    unsigned short *Woth = (unsigned short *)(p + off); off += WOT;
    unsigned short *Wotl = (unsigned short *)(p + off); off += WOT;
    unsigned short *Qhi  = (unsigned short *)(p + off); off += HPLANE;
    unsigned short *Qlo  = (unsigned short *)(p + off); off += HPLANE;
    unsigned short *Khi  = (unsigned short *)(p + off); off += HPLANE;
    unsigned short *Klo  = (unsigned short *)(p + off); off += HPLANE;
    unsigned short *Vthi = (unsigned short *)(p + off); off += HPLANE;
    unsigned short *Vtlo = (unsigned short *)(p + off); off += HPLANE;
    unsigned short *Ohi  = (unsigned short *)(p + off); off += HPLANE;
    unsigned short *Olo  = (unsigned short *)(p + off); off += HPLANE;

    // 0) precision-split / transpose prep
    {
        int n = Mrows * Dc;
        k_split<<<(n + 255) / 256, 256, 0, stream>>>(x, Xhi, Xlo, n);
    }
    {
        int n = Dc * Nqkv;
        k_transpose_split<<<(n + 255) / 256, 256, 0, stream>>>(Wqkv, Wqth, Wqtl, Dc, Nqkv);
    }
    {
        int n = Dc * Dc;
        k_transpose_split<<<(n + 255) / 256, 256, 0, stream>>>(Wout, Woth, Wotl, Dc, Dc);
    }
    // 1) QKV projection
    k_qkv_gemm<<<dim3(Nqkv / 128, Mrows / 128), 256, 0, stream>>>(
        Xhi, Xlo, Wqth, Wqtl, bqkv, Qhi, Qlo, Khi, Klo, Vthi, Vtlo);
    // 2) flash attention
    k_attn<<<dim3(Sc / 64, Hc, Bc), 128, 0, stream>>>(
        Qhi, Qlo, Khi, Klo, Vthi, Vtlo, vlen, Ohi, Olo);
    // 3) output projection
    k_out_gemm<<<dim3(Dc / 128, Mrows / 128), 256, 0, stream>>>(
        Ohi, Olo, Woth, Wotl, bout, out);
}
